// DeltaModel_22007412424941
// MI455X (gfx1250) — matrix operand order repaired and checked by reading, not compiled, not run on hardware
//
#include <hip/hip_runtime.h>

// ---------------------------------------------------------------------------
// Problem constants (match reference)
// ---------------------------------------------------------------------------
static constexpr int N   = 50000;
static constexpr int E   = 800000;
static constexpr int G   = 100;
static constexpr int FIN = 74;
static constexpr int H   = 256;
static constexpr int KP0 = 96;     // FIN padded up to multiple of 32

typedef __attribute__((ext_vector_type(16))) __bf16 v16bf;
typedef __attribute__((ext_vector_type(8)))  float  v8f;

union BF16x16 {
    uint4  q[2];
    v16bf  v;
};

__device__ __forceinline__ unsigned short f32_to_bf16(float f) {
    unsigned int u = __float_as_uint(f);
    unsigned int r = 0x7FFFu + ((u >> 16) & 1u);   // round-to-nearest-even
    return (unsigned short)((u + r) >> 16);
}
__device__ __forceinline__ float bf16_to_f32(unsigned short h) {
    return __uint_as_float(((unsigned int)h) << 16);
}

// ---------------------------------------------------------------------------
// Utility: zero a u32 region
// ---------------------------------------------------------------------------
__global__ void k_zero_u32(unsigned int* __restrict__ p, long long n) {
    long long t = (long long)blockIdx.x * blockDim.x + threadIdx.x;
    if (t < n) p[t] = 0u;
}

// ---------------------------------------------------------------------------
// Degree accumulation + rsqrt(max(deg,1)) transform
// ---------------------------------------------------------------------------
__global__ void k_deg_accum(const int* __restrict__ src, const int* __restrict__ dst,
                            float* __restrict__ dout, float* __restrict__ din, int e) {
    int t = blockIdx.x * blockDim.x + threadIdx.x;
    if (t >= e) return;
    atomicAdd(&dout[src[t]], 1.0f);
    atomicAdd(&din[dst[t]], 1.0f);
}

__global__ void k_isqrt_inplace(float* __restrict__ v, int n) {
    int t = blockIdx.x * blockDim.x + threadIdx.x;
    if (t < n) v[t] = rsqrtf(fmaxf(v[t], 1.0f));
}

// ---------------------------------------------------------------------------
// Layer-0 input prep: Xb[i, 0:KP0] = bf16(x[i, j] * do_isq[i]) (zero padded)
// ---------------------------------------------------------------------------
__global__ void k_prep_x0(const float* __restrict__ x, const float* __restrict__ dox,
                          unsigned short* __restrict__ Xb, int n) {
    long long t = (long long)blockIdx.x * blockDim.x + threadIdx.x;
    if (t >= (long long)n * KP0) return;
    int i = (int)(t / KP0), j = (int)(t % KP0);
    float v = (j < FIN) ? x[(size_t)i * FIN + j] * dox[i] : 0.0f;
    Xb[t] = f32_to_bf16(v);
}

// ---------------------------------------------------------------------------
// Weight packing into WMMA B-operand layout:
//   Wp[((kt*256 + n)*16) + i] = bf16(W[kt*16 + i, n]) for k < K else 0
// so a lane's 16 B-elements (K = ktile16 rows, fixed column n) are contiguous.
// ---------------------------------------------------------------------------
__global__ void k_prep_w(const float* __restrict__ W, unsigned short* __restrict__ Wp,
                         int K, int KP) {
    long long total = (long long)(KP / 16) * 256 * 16;
    long long t = (long long)blockIdx.x * blockDim.x + threadIdx.x;
    if (t >= total) return;
    int i  = (int)(t & 15);
    int n  = (int)((t >> 4) & 255);
    int kt = (int)(t >> 12);
    int k  = kt * 16 + i;
    Wp[t] = (k < K) ? f32_to_bf16(W[(size_t)k * 256 + n]) : (unsigned short)0;
}

// ---------------------------------------------------------------------------
// GEMM: Out_bf16[rows,256] = Xb[rows,KP] @ Wp   (bf16 in, f32 WMMA accumulate)
// One wave32 computes a 16(M) x 64(N) tile: A loaded once per K-step, then
// 4 independent v_wmma_f32_16x16x32_bf16 issues (XDL pipe stays busy, A-panel
// L2 traffic cut 4x vs one-tile-per-wave).
// ---------------------------------------------------------------------------
__global__ void k_gemm_wmma_bf16(const unsigned short* __restrict__ Xb,
                                 const unsigned short* __restrict__ Wp,
                                 unsigned short* __restrict__ Out,
                                 int rows, int KP) {
    int wid  = blockIdx.x * (blockDim.x >> 5) + (threadIdx.x >> 5);
    int lane = threadIdx.x & 31;
    int mt = wid >> 2;        // M tile index (16 rows each)
    int ng = wid & 3;         // N group index (64 cols each)
    if (mt * 16 >= rows) return;

    int l16   = lane & 15;
    int half  = lane >> 4;
    int row   = mt * 16 + l16;
    int nbase = ng * 64;

    v8f c0 = {}, c1 = {}, c2 = {}, c3 = {};
    for (int kk = 0; kk < KP; kk += 32) {
        // A: 16x32 bf16. lane(l16,half) holds K = kk + half*16 + [0..16)
        BF16x16 a;
        const uint4* pa = (const uint4*)(Xb + (size_t)row * KP + kk + half * 16);
        a.q[0] = pa[0];   // 8 bf16 @ kk + half*16
        a.q[1] = pa[1];   // 8 bf16 @ kk + half*16 + 8

        // B: 32x16 bf16 per sub-tile. lane holds W[kk + half*16 + i, n], i=0..15
        size_t brow = ((size_t)(kk / 16 + half) * 256);
        #pragma unroll
        for (int u = 0; u < 4; ++u) {
            BF16x16 b;
            const uint4* pb = (const uint4*)(Wp +
                (brow + (size_t)(nbase + u * 16 + l16)) * 16);
            b.q[0] = pb[0];
            b.q[1] = pb[1];
            v8f& c = (u == 0) ? c0 : (u == 1) ? c1 : (u == 2) ? c2 : c3;
            c = __builtin_amdgcn_wmma_f32_16x16x32_bf16(
                    false, a.v, false, b.v, (short)0, c, false, false);
        }
    }

    // D layout: VGPR r, lanes 0-15 -> M=r, lanes 16-31 -> M=8+r; N = l16
    int mbase = mt * 16 + half * 8;
    #pragma unroll
    for (int u = 0; u < 4; ++u) {
        const v8f& c = (u == 0) ? c0 : (u == 1) ? c1 : (u == 2) ? c2 : c3;
        int coln = nbase + u * 16 + l16;
        #pragma unroll
        for (int r = 0; r < 8; ++r)
            Out[(size_t)(mbase + r) * 256 + coln] = f32_to_bf16(c[r]);
    }
}

// ---------------------------------------------------------------------------
// Edge scatter-add: Agg[dst] += Tb[src], bf16 features, packed bf16 atomics.
// One thread handles 8 features (one 16B gather, 4 pk_add_bf16 atomics).
// A wave covers one edge (32 groups x 8 feats = 256): index loads are
// wave-uniform (single cacheline), row gathers are fully coalesced (512B).
// ---------------------------------------------------------------------------
__device__ __forceinline__ void atomic_pk_add_bf16(unsigned long long addr,
                                                   unsigned int data) {
    asm volatile("global_atomic_pk_add_bf16 %0, %1, off"
                 :: "v"(addr), "v"(data) : "memory");
}

__global__ void k_scatter_pk(const unsigned short* __restrict__ Tb,
                             const int* __restrict__ src, const int* __restrict__ dst,
                             unsigned short* __restrict__ Agg, int e) {
    long long t = (long long)blockIdx.x * blockDim.x + threadIdx.x;
    int grp = (int)(t & 31);           // 32 groups x 8 feats = 256
    long long ed = t >> 5;
    if (ed >= e) return;
    int s = src[ed], d = dst[ed];
    uint4 v = *(const uint4*)(Tb + (size_t)s * 256 + grp * 8);
    unsigned long long base =
        (unsigned long long)(Agg + (size_t)d * 256 + grp * 8);
    const unsigned int* w = (const unsigned int*)&v;
    #pragma unroll
    for (int j = 0; j < 4; ++j)
        atomic_pk_add_bf16(base + (unsigned long long)j * 4ull, w[j]);
}

// ---------------------------------------------------------------------------
// Post-aggregation elementwise: y = relu(Agg*di + b)
//   Xb_next = bf16(y * do)  (input of next layer, pre-scaled)
//   Hb      = bf16(y)       (last layer: pooling input)
// ---------------------------------------------------------------------------
__global__ void k_post(const unsigned short* __restrict__ Agg,
                       const float* __restrict__ di, const float* __restrict__ dox,
                       const float* __restrict__ bias,
                       unsigned short* __restrict__ Xb_next,
                       unsigned short* __restrict__ Hb, int n) {
    long long t = (long long)blockIdx.x * blockDim.x + threadIdx.x;
    if (t >= (long long)n * 256) return;
    int i = (int)(t >> 8), j = (int)(t & 255);
    float y = fmaxf(bf16_to_f32(Agg[t]) * di[i] + bias[j], 0.0f);
    if (Xb_next) Xb_next[t] = f32_to_bf16(y * dox[i]);
    if (Hb)      Hb[t]      = f32_to_bf16(y);
}

// ---------------------------------------------------------------------------
// Per-graph pooling: P[gid[i], j] += Hb[i, j]  (f32 accumulate)
// ---------------------------------------------------------------------------
__global__ void k_pool(const unsigned short* __restrict__ Hb,
                       const int* __restrict__ gid, float* __restrict__ P, int n) {
    long long t = (long long)blockIdx.x * blockDim.x + threadIdx.x;
    if (t >= (long long)n * 256) return;
    int i = (int)(t >> 8), j = (int)(t & 255);
    atomicAdd(&P[(size_t)gid[i] * 256 + j], bf16_to_f32(Hb[t]));
}

// ---------------------------------------------------------------------------
// Head MLP: out[g] = relu([Pl|Pr] @ Wf1 + bf1) @ Wf2 + bf2   (tiny, f32)
// One block per graph, 256 threads: thread j owns hidden unit j.
// ---------------------------------------------------------------------------
__global__ void k_head(const float* __restrict__ Pl, const float* __restrict__ Pr,
                       const float* __restrict__ Wf1, const float* __restrict__ bf1,
                       const float* __restrict__ Wf2, const float* __restrict__ bf2,
                       float* __restrict__ out) {
    int g = blockIdx.x, j = threadIdx.x;
    float acc = bf1[j];
    for (int k = 0; k < 256; ++k)
        acc += Pl[(size_t)g * 256 + k] * Wf1[(size_t)k * 256 + j];
    for (int k = 0; k < 256; ++k)
        acc += Pr[(size_t)g * 256 + k] * Wf1[(size_t)(256 + k) * 256 + j];
    float h = fmaxf(acc, 0.0f) * Wf2[j];
    __shared__ float red[256];
    red[j] = h;
    __syncthreads();
    for (int s = 128; s > 0; s >>= 1) {
        if (j < s) red[j] += red[j + s];
        __syncthreads();
    }
    if (j == 0) out[g] = red[0] + bf2[0];
}

// ---------------------------------------------------------------------------
// Host driver
// ---------------------------------------------------------------------------
extern "C" void kernel_launch(void* const* d_in, const int* in_sizes, int n_in,
                              void* d_out, int out_size, void* d_ws, size_t ws_size,
                              hipStream_t stream) {
    const float* xl    = (const float*)d_in[0];
    const float* xr    = (const float*)d_in[1];
    const int*   src_l = (const int*)d_in[2];
    const int*   dst_l = (const int*)d_in[3];
    const int*   src_r = (const int*)d_in[4];
    const int*   dst_r = (const int*)d_in[5];
    const int*   gid_l = (const int*)d_in[6];
    const int*   gid_r = (const int*)d_in[7];
    const float* W0    = (const float*)d_in[8];
    const float* b0    = (const float*)d_in[9];
    const float* W1    = (const float*)d_in[10];
    const float* b1    = (const float*)d_in[11];
    const float* W2    = (const float*)d_in[12];
    const float* b2    = (const float*)d_in[13];
    const float* Wf1   = (const float*)d_in[14];
    const float* bf1   = (const float*)d_in[15];
    const float* Wf2   = (const float*)d_in[16];
    const float* bf2   = (const float*)d_in[17];
    float* out = (float*)d_out;

    // --- workspace carve-up (256B aligned) ---
    size_t off = 0;
    auto alloc = [&](size_t bytes) -> void* {
        void* p = (char*)d_ws + off;
        off += (bytes + 255) & ~(size_t)255;
        return p;
    };
    float* do_l = (float*)alloc((size_t)N * 4);
    float* di_l = (float*)alloc((size_t)N * 4);
    float* do_r = (float*)alloc((size_t)N * 4);
    float* di_r = (float*)alloc((size_t)N * 4);
    unsigned short* Xb_l  = (unsigned short*)alloc((size_t)N * 256 * 2);
    unsigned short* Xb_r  = (unsigned short*)alloc((size_t)N * 256 * 2);
    unsigned short* Tb_l  = (unsigned short*)alloc((size_t)N * 256 * 2);
    unsigned short* Tb_r  = (unsigned short*)alloc((size_t)N * 256 * 2);
    unsigned short* Agg_l = (unsigned short*)alloc((size_t)N * 256 * 2);
    unsigned short* Agg_r = (unsigned short*)alloc((size_t)N * 256 * 2);
    unsigned short* Wp0 = (unsigned short*)alloc((size_t)KP0 * 256 * 2);
    unsigned short* Wp1 = (unsigned short*)alloc((size_t)256 * 256 * 2);
    unsigned short* Wp2 = (unsigned short*)alloc((size_t)256 * 256 * 2);
    float* Pl = (float*)alloc((size_t)G * 256 * 4);
    float* Pr = (float*)alloc((size_t)G * 256 * 4);

    auto zero = [&](void* p, long long u32_count) {
        int blk = 256;
        long long grid = (u32_count + blk - 1) / blk;
        k_zero_u32<<<(unsigned)grid, blk, 0, stream>>>((unsigned int*)p, u32_count);
    };

    // --- degrees -> rsqrt scales ---
    zero(do_l, N); zero(di_l, N); zero(do_r, N); zero(di_r, N);
    {
        int blk = 256, grid = (E + blk - 1) / blk;
        k_deg_accum<<<grid, blk, 0, stream>>>(src_l, dst_l, do_l, di_l, E);
        k_deg_accum<<<grid, blk, 0, stream>>>(src_r, dst_r, do_r, di_r, E);
        int g2 = (N + blk - 1) / blk;
        k_isqrt_inplace<<<g2, blk, 0, stream>>>(do_l, N);
        k_isqrt_inplace<<<g2, blk, 0, stream>>>(di_l, N);
        k_isqrt_inplace<<<g2, blk, 0, stream>>>(do_r, N);
        k_isqrt_inplace<<<g2, blk, 0, stream>>>(di_r, N);
    }

    // --- weight packing ---
    {
        int blk = 256;
        long long t0 = (long long)(KP0 / 16) * 256 * 16;
        long long t1 = (long long)(256 / 16) * 256 * 16;
        k_prep_w<<<(unsigned)((t0 + blk - 1) / blk), blk, 0, stream>>>(W0, Wp0, FIN, KP0);
        k_prep_w<<<(unsigned)((t1 + blk - 1) / blk), blk, 0, stream>>>(W1, Wp1, 256, 256);
        k_prep_w<<<(unsigned)((t1 + blk - 1) / blk), blk, 0, stream>>>(W2, Wp2, 256, 256);
    }

    // --- layer-0 input prep (fold D_out^{-1/2} scaling into bf16 cast) ---
    {
        int blk = 256;
        long long t = (long long)N * KP0;
        k_prep_x0<<<(unsigned)((t + blk - 1) / blk), blk, 0, stream>>>(xl, do_l, Xb_l, N);
        k_prep_x0<<<(unsigned)((t + blk - 1) / blk), blk, 0, stream>>>(xr, do_r, Xb_r, N);
    }

    const unsigned short* Wps[3]    = {Wp0, Wp1, Wp2};
    const float*          biases[3] = {b0, b1, b2};
    const int             KPs[3]    = {KP0, 256, 256};

    struct Stream {
        const int *src, *dst;
        const float *dox, *di;
        unsigned short *Xb, *Tb, *Agg;
    } S[2] = {
        { src_l, dst_l, do_l, di_l, Xb_l, Tb_l, Agg_l },
        { src_r, dst_r, do_r, di_r, Xb_r, Tb_r, Agg_r },
    };

    const int gemm_waves  = (N / 16) * 4;                  // 12500 waves (16x64 tiles)
    const int gemm_blocks = (gemm_waves + 7) / 8;          // 8 waves / block
    const long long scat_threads = (long long)E * 32;
    const unsigned scat_blocks = (unsigned)((scat_threads + 255) / 256);
    const long long elem = (long long)N * 256;
    const unsigned elem_blocks = (unsigned)((elem + 255) / 256);

    for (int layer = 0; layer < 3; ++layer) {
        for (int s = 0; s < 2; ++s) {
            // 1) T = Xb @ W  (WMMA bf16, f32 acc, bf16 out)
            k_gemm_wmma_bf16<<<gemm_blocks, 256, 0, stream>>>(
                S[s].Xb, Wps[layer], S[s].Tb, N, KPs[layer]);
            // 2) Agg = scatter-add over edges (packed bf16 atomics)
            zero(S[s].Agg, (long long)N * 128);  // N*256 bf16 == N*128 u32
            k_scatter_pk<<<scat_blocks, 256, 0, stream>>>(
                S[s].Tb, S[s].src, S[s].dst, S[s].Agg, E);
            // 3) relu(Agg*di + b); write next-layer input or pooling input
            if (layer < 2) {
                k_post<<<elem_blocks, 256, 0, stream>>>(
                    S[s].Agg, S[s].di, S[s].dox, biases[layer],
                    S[s].Xb, (unsigned short*)nullptr, N);
            } else {
                k_post<<<elem_blocks, 256, 0, stream>>>(
                    S[s].Agg, S[s].di, S[s].dox, biases[layer],
                    (unsigned short*)nullptr, S[s].Tb, N);
            }
        }
    }

    // --- per-graph pooling ---
    zero(Pl, (long long)G * 256);
    zero(Pr, (long long)G * 256);
    k_pool<<<elem_blocks, 256, 0, stream>>>(Tb_l, gid_l, Pl, N);
    k_pool<<<elem_blocks, 256, 0, stream>>>(Tb_r, gid_r, Pr, N);

    // --- head MLP ---
    k_head<<<G, 256, 0, stream>>>(Pl, Pr, Wf1, bf1, Wf2, bf2, out);

    (void)in_sizes; (void)n_in; (void)out_size; (void)ws_size;
}
